// SampleNetMLP_36189394436483
// MI455X (gfx1250) — matrix operand order repaired and checked by reading, not compiled, not run on hardware
//
#include <hip/hip_runtime.h>
#include <hip/hip_bf16.h>
#include <stdint.h>

// Problem constants (match reference)
#define BB 2
#define CC 64
#define HH 96
#define WW 96
#define QQ 65536
#define IN_DIM 580
#define K0 608              // layer-0 K padded to multiple of 32
#define HID 256
#define QT 16               // queries per block
#define NQT (QQ / QT)       // 4096 query tiles per batch image
#define NBLK (BB * NQT)     // 8192 blocks

typedef __attribute__((ext_vector_type(16))) _Float16 v16h;
typedef __attribute__((ext_vector_type(8)))  float    v8f;

// d_ws element offsets (in _Float16 units) for converted weights
#define OFF_W0 0
#define OFF_W1 (OFF_W0 + 256 * K0)       // 155648
#define OFF_W2 (OFF_W1 + 256 * 256)
#define OFF_W3 (OFF_W2 + 256 * 256)
#define OFF_W4 (OFF_W3 + 256 * 256)
#define OFF_END (OFF_W4 + 16 * 256)      // 356352 halves
#define WS_PART_BYTES (OFF_END * 2)      // partials (float) start here

// Dynamic LDS pool layout (bytes)
#define SH_IN   (QT * K0 * 2)            // 19456 per shift
#define SH_HID  (QT * HID * 2)           // 8192  per shift
#define POOL0_BYTES (4 * SH_IN)          // 77824: in_buf[4]; hA overlays after L0
#define POOL1_BYTES (4 * SH_HID)         // 32768: hB[4]
#define POOL_BYTES  (POOL0_BYTES + POOL1_BYTES)   // 110592

// ---------------------------------------------------------------------------
// Weight conversion: fp32 (K-major, [k][n]) -> f16 N-major [n][k], K padded.
// ---------------------------------------------------------------------------
__global__ void conv_weights(const float* __restrict__ w0,
                             const float* __restrict__ w1,
                             const float* __restrict__ w2,
                             const float* __restrict__ w3,
                             const float* __restrict__ w4,
                             _Float16* __restrict__ ws) {
  const int n0 = 256 * K0;
  const int nh = 256 * 256;
  const int n4 = 16 * 256;
  const int total = n0 + 3 * nh + n4;
  for (int idx = blockIdx.x * blockDim.x + threadIdx.x; idx < total;
       idx += gridDim.x * blockDim.x) {
    if (idx < n0) {
      int n = idx / K0, k = idx % K0;
      ws[OFF_W0 + idx] =
          (k < IN_DIM) ? (_Float16)w0[k * 256 + n] : (_Float16)0.0f;
    } else if (idx < n0 + 3 * nh) {
      int j = idx - n0;
      int l = j / nh, r = j % nh;
      int n = r / 256, k = r % 256;
      const float* w = (l == 0) ? w1 : ((l == 1) ? w2 : w3);
      ws[OFF_W1 + j] = (_Float16)w[k * 256 + n];
    } else {
      int j = idx - n0 - 3 * nh;
      int n = j / 256, k = j % 256;
      ws[OFF_W4 + j] = (n < 2) ? (_Float16)w4[k * 2 + n] : (_Float16)0.0f;
    }
  }
}

// ---------------------------------------------------------------------------
// Fragment loaders (wave32; A and B both use contiguous k = 16*half + s)
// ---------------------------------------------------------------------------
// A 16x32 f16: lane holds row m=lane&15; half=lane>>4 selects the K subset.
// elements 0..15 = K half*16 .. half*16+15 (relative to
// row + half*16): two contiguous 16B spans -> 2x ds_load_b128.
__device__ __forceinline__ v16h load_a(const _Float16* row) {
  union { v16h v; uint4 q[2]; } f;
  f.q[0] = *(const uint4*)(row);
  f.q[1] = *(const uint4*)(row + 8);
  return f.v;
}

// B 32x16 f16: lane holds column n=lane&15; half selects K 0-15 / 16-31.
// 16 contiguous f16 (32B) per lane.
__device__ __forceinline__ v16h load_b(const _Float16* p) {
  union { v16h v; uint4 q[2]; } f;
  const uint4* g = (const uint4*)p;
  f.q[0] = g[0];
  f.q[1] = g[1];
  return f.v;
}

// One dense layer over ALL FOUR shifts: wave owns N-tiles {2w, 2w+1}.
// Each weight fragment is loaded once and consumed by 4 WMMAs, one per shift.
__device__ __forceinline__ void mlp_layer4s(const _Float16* __restrict__ inb,
                                            int in_shift_stride,  // elements
                                            int kpad, int kc_count,
                                            const _Float16* __restrict__ Wg,
                                            const float* __restrict__ bias,
                                            _Float16* __restrict__ outb,
                                            int out_shift_stride, // elements
                                            int wave, int lane) {
  const int m = lane & 15;
  const int half = lane >> 4;
  const int nt0 = wave * 2, nt1 = nt0 + 1;
  v8f acc[4][2];
#pragma unroll
  for (int s = 0; s < 4; ++s) {
    acc[s][0] = {};
    acc[s][1] = {};
  }
#pragma unroll 1
  for (int kc = 0; kc < kc_count; ++kc) {
    const int kb = kc * 32;
    const _Float16* wb = Wg + kb + half * 16;
    v16h B0 = load_b(wb + (nt0 * 16 + m) * kpad);
    v16h B1 = load_b(wb + (nt1 * 16 + m) * kpad);
#pragma unroll
    for (int s = 0; s < 4; ++s) {
      v16h A = load_a(inb + s * in_shift_stride + m * kpad + kb + half * 16);
      acc[s][0] = __builtin_amdgcn_wmma_f32_16x16x32_f16(
          false, A, false, B0, (short)0, acc[s][0], false, false);
      acc[s][1] = __builtin_amdgcn_wmma_f32_16x16x32_f16(
          false, A, false, B1, (short)0, acc[s][1], false, false);
    }
  }
  const float bb0 = bias[nt0 * 16 + m];
  const float bb1 = bias[nt1 * 16 + m];
#pragma unroll
  for (int s = 0; s < 4; ++s) {
    _Float16* ob = outb + s * out_shift_stride;
#pragma unroll
    for (int v = 0; v < 8; ++v) {
      int row = v + 8 * half;  // D layout: lane=col, VGPR v = row v+8*half
      float x0 = acc[s][0][v] + bb0; x0 = x0 > 0.0f ? x0 : 0.0f;
      float x1 = acc[s][1][v] + bb1; x1 = x1 > 0.0f ? x1 : 0.0f;
      ob[row * HID + nt0 * 16 + m] = (_Float16)x0;
      ob[row * HID + nt1 * 16 + m] = (_Float16)x1;
    }
  }
}

// ---------------------------------------------------------------------------
// Fused LIIF kernel: one block = 16 queries; all 4 shifted taps processed
// together so every weight fragment is loaded exactly once per block.
// ---------------------------------------------------------------------------
__global__ __launch_bounds__(256) void liif_kernel(
    const float* __restrict__ feat, const float* __restrict__ coord,
    const float* __restrict__ cell,
    const float* __restrict__ b0, const float* __restrict__ b1,
    const float* __restrict__ b2, const float* __restrict__ b3,
    const float* __restrict__ b4,
    const _Float16* __restrict__ ws,
    float* __restrict__ out, float* __restrict__ partial) {
  extern __shared__ char pool[];
  _Float16* in_buf = (_Float16*)pool;                  // [4][QT*K0]
  _Float16* hA = (_Float16*)pool;                      // [4][QT*HID] overlay
  _Float16* hB = (_Float16*)(pool + POOL0_BYTES);      // [4][QT*HID]

  __shared__ int   s_base[4][QT], s_mask[4][QT];
  __shared__ float s_rcx[4][QT], s_rcy[4][QT];
  __shared__ float s_area[4][QT], s_wgt[4][QT];
  __shared__ float s_clx[QT], s_cly[QT];

  const int t = threadIdx.x;
  const int wave = t >> 5;
  const int lane = t & 31;
  const int blk = blockIdx.x;
  const int b = blk / NQT;
  const int q0 = (blk % NQT) * QT;

  const _Float16* W0 = ws + OFF_W0;
  const _Float16* W1 = ws + OFF_W1;
  const _Float16* W2 = ws + OFF_W2;
  const _Float16* W3 = ws + OFF_W3;
  const _Float16* W4 = ws + OFF_W4;

  // --- per-(shift,query) geometry: base offset, 3x3 bounds mask, areas ---
  if (t < 64) {
    const int s = t >> 4, q = t & 15;
    const int qg = q0 + q;
    const float cx = coord[(b * QQ + qg) * 2 + 0];
    const float cy = coord[(b * QQ + qg) * 2 + 1];
    const float vx = (s & 2) ? 1.0f : -1.0f;
    const float vy = (s & 1) ? 1.0f : -1.0f;
    const float eps = 1e-6f;
    float c0 = fminf(fmaxf(cx + vx * (1.0f / HH) + eps, -1.0f + eps), 1.0f - eps);
    float c1 = fminf(fmaxf(cy + vy * (1.0f / WW) + eps, -1.0f + eps), 1.0f - eps);
    int iy = (int)floorf(((c0 + 1.0f) * (float)HH - 1.0f) * 0.5f + 0.5f);
    int ix = (int)floorf(((c1 + 1.0f) * (float)WW - 1.0f) * 0.5f + 0.5f);
    iy = iy < 0 ? 0 : (iy > HH - 1 ? HH - 1 : iy);
    ix = ix < 0 ? 0 : (ix > WW - 1 ? WW - 1 : ix);
    const float qcx = -1.0f + (2.0f * (float)iy + 1.0f) / (float)HH;
    const float qcy = -1.0f + (2.0f * (float)ix + 1.0f) / (float)WW;
    const float rcx = (cx - qcx) * (float)HH;
    const float rcy = (cy - qcy) * (float)WW;
    int mask = 0;
#pragma unroll
    for (int kk = 0; kk < 9; ++kk) {
      const int y = iy + kk / 3 - 1;
      const int x = ix + kk % 3 - 1;
      if (y >= 0 && y < HH && x >= 0 && x < WW) mask |= 1 << kk;
    }
    s_base[s][q] = (iy - 1) * WW + (ix - 1);
    s_mask[s][q] = mask;
    s_rcx[s][q] = rcx; s_rcy[s][q] = rcy;
    s_area[s][q] = fabsf(rcx * rcy) + 1e-9f;
    if (s == 0) {
      s_clx[q] = cell[(b * QQ + qg) * 2 + 0] * (float)HH;
      s_cly[q] = cell[(b * QQ + qg) * 2 + 1] * (float)WW;
    }
  }
  // zero the K padding region [IN_DIM, K0) for all 4 shifts
  for (int idx = t; idx < 4 * QT * (K0 - IN_DIM); idx += 256) {
    int s = idx / (QT * (K0 - IN_DIM));
    int r = idx % (QT * (K0 - IN_DIM));
    int q = r / (K0 - IN_DIM);
    int d = IN_DIM + r % (K0 - IN_DIM);
    in_buf[(s * QT + q) * K0 + d] = (_Float16)0.0f;
  }
  __syncthreads();
  if (t < QT) {  // blend weights with diagonal area swap
    float tot = s_area[0][t] + s_area[1][t] + s_area[2][t] + s_area[3][t];
#pragma unroll
    for (int s = 0; s < 4; ++s) s_wgt[s][t] = s_area[3 - s][t] / tot;
  }

  // --- gather all four 580-dim input tiles into LDS (f16, packed u32) ---
  // Thread (q = t>>4, ln = t&15) handles element pairs d = 2*ln + 32*j.
  // Incremental (c, kk): d += 32  ==>  c += 3, kk += 5 (wrap at 9).
  {
    const int q = t >> 4;
    const int ln = t & 15;
    const float* featb = feat + (size_t)b * CC * HH * WW;
    const int d0 = 2 * ln;
    const int c0i = d0 / 9;
    const int kk0 = d0 - 9 * c0i;
#pragma unroll 1
    for (int s = 0; s < 4; ++s) {
      const int base = s_base[s][q];
      const int mask = s_mask[s][q];
      _Float16* dst = in_buf + (s * QT + q) * K0;
      int c = c0i, kk = kk0;
#pragma unroll 1
      for (int d = d0; d < IN_DIM; d += 32) {
        float v0, v1;
        if (d < 576) {
          const int ki = kk / 3, kj = kk - 3 * ki;
          v0 = ((mask >> kk) & 1)
                   ? featb[c * (HH * WW) + base + ki * WW + kj] : 0.0f;
          int c1 = c, kk1 = kk + 1;
          if (kk1 == 9) { kk1 = 0; c1 = c + 1; }
          const int ki1 = kk1 / 3, kj1 = kk1 - 3 * ki1;
          v1 = ((mask >> kk1) & 1)
                   ? featb[c1 * (HH * WW) + base + ki1 * WW + kj1] : 0.0f;
        } else if (d == 576) {
          v0 = s_rcx[s][q]; v1 = s_rcy[s][q];
        } else {  // d == 578
          v0 = s_clx[q];    v1 = s_cly[q];
        }
        union { _Float16 h[2]; uint32_t u; } pk;
        pk.h[0] = (_Float16)v0;
        pk.h[1] = (_Float16)v1;
        *(uint32_t*)(dst + d) = pk.u;
        c += 3; kk += 5;
        if (kk >= 9) { kk -= 9; c += 1; }
      }
    }
  }
  __syncthreads();

  // --- MLP over all 4 shifts (WMMA, 8 waves split N) ---
  mlp_layer4s(in_buf, QT * K0, K0, K0 / 32, W0, b0, hB, QT * HID, wave, lane);
  __syncthreads();   // hA overlays in_buf from here on
  mlp_layer4s(hB, QT * HID, HID, HID / 32, W1, b1, hA, QT * HID, wave, lane);
  __syncthreads();
  mlp_layer4s(hA, QT * HID, HID, HID / 32, W2, b2, hB, QT * HID, wave, lane);
  __syncthreads();
  mlp_layer4s(hB, QT * HID, HID, HID / 32, W3, b3, hA, QT * HID, wave, lane);
  __syncthreads();

  // --- final 256->2 (N padded to 16) on wave 0; blend all 4 shifts ---
  if (wave == 0) {
    const int m = lane & 15, half = lane >> 4;
    v8f p[4];
#pragma unroll
    for (int s = 0; s < 4; ++s) p[s] = {};
#pragma unroll 1
    for (int kc = 0; kc < HID / 32; ++kc) {
      const int kb = kc * 32;
      v16h Bf = load_b(W4 + m * HID + kb + half * 16);
#pragma unroll
      for (int s = 0; s < 4; ++s) {
        v16h A = load_a(hA + s * QT * HID + m * HID + kb + half * 16);
        p[s] = __builtin_amdgcn_wmma_f32_16x16x32_f16(false, A, false, Bf,
                                                      (short)0, p[s], false,
                                                      false);
      }
    }
    const float bv = (m < 2) ? b4[m] : 0.0f;
    v8f ret = {};
#pragma unroll
    for (int s = 0; s < 4; ++s) {
#pragma unroll
      for (int v = 0; v < 8; ++v) {
        const int row = v + 8 * half;
        ret[v] += (p[s][v] + bv) * s_wgt[s][row];
      }
    }
    // write ret (B,Q,2) and per-block |ret| partial sum
    const int n = m;
    float asum = 0.0f;
    if (n < 2) {
#pragma unroll
      for (int v = 0; v < 8; ++v) {
        const int row = v + 8 * half;
        out[(b * QQ + q0 + row) * 2 + n] = ret[v];
        asum += fabsf(ret[v]);
      }
    }
#pragma unroll
    for (int off = 16; off > 0; off >>= 1)
      asum += __shfl_xor(asum, off, 32);
    if (lane == 0) partial[blk] = asum;
  }
}

// ---------------------------------------------------------------------------
// Deterministic final reduction: mean(|ret|) -> d_out[B*Q*2]
// ---------------------------------------------------------------------------
__global__ void reduce_abs(const float* __restrict__ partial,
                           float* __restrict__ out_scalar) {
  __shared__ float sm[256];
  float s = 0.0f;
  for (int i = threadIdx.x; i < NBLK; i += 256) s += partial[i];
  sm[threadIdx.x] = s;
  __syncthreads();
  for (int o = 128; o > 0; o >>= 1) {
    if (threadIdx.x < o) sm[threadIdx.x] += sm[threadIdx.x + o];
    __syncthreads();
  }
  if (threadIdx.x == 0)
    out_scalar[0] = sm[0] / (float)(BB * QQ * 2);
}

extern "C" void kernel_launch(void* const* d_in, const int* in_sizes, int n_in,
                              void* d_out, int out_size, void* d_ws,
                              size_t ws_size, hipStream_t stream) {
  (void)in_sizes; (void)n_in; (void)out_size; (void)ws_size;
  const float* feat  = (const float*)d_in[0];
  const float* coord = (const float*)d_in[1];
  const float* cell  = (const float*)d_in[2];
  const float* w0 = (const float*)d_in[3];
  const float* b0 = (const float*)d_in[4];
  const float* w1 = (const float*)d_in[5];
  const float* b1 = (const float*)d_in[6];
  const float* w2 = (const float*)d_in[7];
  const float* b2 = (const float*)d_in[8];
  const float* w3 = (const float*)d_in[9];
  const float* b3 = (const float*)d_in[10];
  const float* w4 = (const float*)d_in[11];
  const float* b4 = (const float*)d_in[12];

  _Float16* ws = (_Float16*)d_ws;
  float* out = (float*)d_out;
  float* partial = (float*)((char*)d_ws + WS_PART_BYTES);

  // Allow >64KB dynamic LDS (gfx1250 WGP has 320KB). Host-side, idempotent,
  // graph-capture safe.
  hipFuncSetAttribute((const void*)liif_kernel,
                      hipFuncAttributeMaxDynamicSharedMemorySize, POOL_BYTES);

  conv_weights<<<512, 256, 0, stream>>>(w0, w1, w2, w3, w4, ws);
  liif_kernel<<<NBLK, 256, POOL_BYTES, stream>>>(feat, coord, cell, b0, b1, b2,
                                                 b3, b4, ws, out, partial);
  reduce_abs<<<1, 256, 0, stream>>>(partial, out + BB * QQ * 2);
}
